// MultiheadMaskedAttention_71133248356497
// MI455X (gfx1250) — matrix operand order repaired and checked by reading, not compiled, not run on hardware
//
#include <hip/hip_runtime.h>
#include <hip/hip_bf16.h>
#include <stdint.h>

// ---------------------------------------------------------------------------
// Problem constants (from reference): B=2, S=2048, H=1024, nh=16, hs=64
// ---------------------------------------------------------------------------
#define BATCH 2
#define SEQ   2048
#define HID   1024
#define NH    16
#define HS    64
#define ROWS  (BATCH * SEQ)          // 4096
#define QKVN  (3 * HID)              // 3072

typedef __attribute__((ext_vector_type(16))) __bf16 v16bf;
typedef __attribute__((ext_vector_type(8)))  float  v8f;
typedef __attribute__((ext_vector_type(4)))  unsigned int u32x4;
typedef __attribute__((ext_vector_type(8)))  unsigned int u32x8;

// ---------------------------------------------------------------------------
// helpers
// ---------------------------------------------------------------------------
static __device__ __forceinline__ __bf16 f2bf(float f) {
  union { float f; uint32_t u; } v; v.f = f;
  uint32_t r = v.u + 0x7FFFu + ((v.u >> 16) & 1u);   // round-to-nearest-even
  unsigned short h = (unsigned short)(r >> 16);
  __bf16 out;
  __builtin_memcpy(&out, &h, 2);
  return out;
}

static __device__ __forceinline__ v8f wmma_bf16(v16bf a, v16bf b, v8f c) {
  // D = A(16x32 bf16) * B(32x16 bf16) + C(16x16 f32)
  return __builtin_amdgcn_wmma_f32_16x16x32_bf16(false, a, false, b,
                                                 (short)0, c, false, false);
}

// ---------------------------------------------------------------------------
// Tensor Data Mover: DMA a 2D tile (tile_d1 rows x tile_d0 elems of 2 bytes)
// from global (row stride d0_stride elems) into LDS at lds_off, inserting
// (pad_amount_code+1) DWORDs of padding every 2^(pad_interval_code+1) DWORDs.
// Issued by one wave; completion tracked in-order via TENSORcnt.
// ---------------------------------------------------------------------------
static __device__ __forceinline__ void tdm_load_2d(
    unsigned int lds_off, const void* gsrc, unsigned int tensor_d0,
    unsigned int tensor_d1, unsigned long long d0_stride, unsigned int tile_d0,
    unsigned int tile_d1, unsigned int pad_interval_code,
    unsigned int pad_amount_code) {
  unsigned long long ga = (unsigned long long)(uintptr_t)gsrc;
  u32x4 g0;
  g0[0] = 1u;                                     // count=1 (valid user D#)
  g0[1] = lds_off;                                // lds_addr
  g0[2] = (unsigned int)ga;                       // global_addr[31:0]
  g0[3] = (unsigned int)(ga >> 32) | (2u << 30);  // global_addr[56:32], type=2
  u32x8 g1;
  g1[0] = (1u << 16)                              // data_size = 2 bytes
        | (1u << 20)                              // pad_enable
        | (pad_interval_code << 22) | (pad_amount_code << 25);
  g1[1] = (tensor_d0 & 0xFFFFu) << 16;            // tensor_dim0[15:0]
  g1[2] = (tensor_d0 >> 16) | ((tensor_d1 & 0xFFFFu) << 16);
  g1[3] = (tensor_d1 >> 16) | ((tile_d0 & 0xFFFFu) << 16);
  g1[4] = tile_d1 & 0xFFFFu;                      // tile_dim1 (tile_dim2=0)
  g1[5] = (unsigned int)d0_stride;                // tensor_dim0_stride[31:0]
  g1[6] = (unsigned int)(d0_stride >> 32) & 0xFFFFu;
  g1[7] = 0u;
  asm volatile("tensor_load_to_lds %0, %1" ::"s"(g0), "s"(g1) : "memory");
}

// A-fragment 16x32 bf16 from LDS (row-major, leading dim ld).
// Lane L holds row M=L%16; element j holds K = 16*(L/16) + j (contiguous).
static __device__ __forceinline__ v16bf load_a16x32(const __bf16* s, int ld,
                                                    int row0, int k0) {
  int lane = threadIdx.x & 31;
  int m = lane & 15, grp = lane >> 4;
  const __bf16* p = s + (size_t)(row0 + m) * ld + k0 + grp * 16;
  v16bf a;
#pragma unroll
  for (int j = 0; j < 16; ++j) {
    int k = j;
    a[j] = p[k];
  }
  return a;
}

// B-fragment 32x16 where LDS stores B-transposed: s is [N][K] row-major.
static __device__ __forceinline__ v16bf load_bT32x16(const __bf16* s, int ld,
                                                     int n0, int k0) {
  int lane = threadIdx.x & 31;
  int n = lane & 15, grp = lane >> 4;
  const __bf16* p = s + (size_t)(n0 + n) * ld + k0 + grp * 16;
  v16bf b;
#pragma unroll
  for (int j = 0; j < 16; ++j) b[j] = p[j];
  return b;
}

// B-fragment 32x16 where LDS stores B row-major: s is [K][N].
static __device__ __forceinline__ v16bf load_b32x16_rm(const __bf16* s, int ld,
                                                       int k0, int n0) {
  int lane = threadIdx.x & 31;
  int n = lane & 15, grp = lane >> 4;
  const __bf16* p = s + (size_t)(k0 + grp * 16) * ld + n0 + n;
  v16bf b;
#pragma unroll
  for (int j = 0; j < 16; ++j) b[j] = p[(size_t)j * ld];
  return b;
}

static __device__ __forceinline__ float redmax16(float x) {
#pragma unroll
  for (int off = 1; off < 16; off <<= 1) x = fmaxf(x, __shfl_xor(x, off, 32));
  return x;
}

static __device__ __forceinline__ unsigned int lds_offset(const void* p) {
  return (unsigned int)(uintptr_t)p;   // generic ptr low 32 bits = LDS offset
}

// ---------------------------------------------------------------------------
// Kernel 1: fp32 -> bf16 conversions (x straight; weights transposed to [N][K])
// ---------------------------------------------------------------------------
__global__ __launch_bounds__(256) void convert_kernel(
    const float* __restrict__ x, const float* __restrict__ Wqkv,
    const float* __restrict__ Wo, __bf16* __restrict__ xb,
    __bf16* __restrict__ wqkvt, __bf16* __restrict__ wot) {
  size_t i = (size_t)blockIdx.x * 256 + threadIdx.x;
  if (i < (size_t)ROWS * HID) xb[i] = f2bf(x[i]);
  if (i < (size_t)QKVN * HID) {
    size_t n = i >> 10, k = i & 1023;               // [n][k] <- W[k][n]
    wqkvt[i] = f2bf(Wqkv[k * QKVN + n]);
  }
  if (i < (size_t)HID * HID) {
    size_t n = i >> 10, k = i & 1023;
    wot[i] = f2bf(Wo[k * HID + n]);
  }
}

// ---------------------------------------------------------------------------
// Kernels 2 & 4: TDM + WMMA GEMM, C = A[M][K] * Bt[N][K]^T + bias
// 128x128 block, K-step 64, 256 threads = 8 waves, wave tile 32x64.
// Double-buffered TDM tiles: 64x2B rows (32 DW -> interval code 4) padded by
// 4 DWORDs (code 3) -> LDS leading dim 72.  DMA of tile i+1 overlaps WMMAs
// of tile i; wave 0 waits tensorcnt<=2 (the two just-issued loads).
// MODE 0: scatter into Q/K/V head-major bf16.  MODE 1: fp32 out + bias.
// ---------------------------------------------------------------------------
#define LDG 72

template <int MODE, int NTOT, int KTOT>
__global__ __launch_bounds__(256) void gemm128_kernel(
    const __bf16* __restrict__ A, const __bf16* __restrict__ Bt,
    const float* __restrict__ bias, __bf16* __restrict__ obf,
    float* __restrict__ of32) {
  __shared__ __align__(16) __bf16 As[2][128 * LDG];
  __shared__ __align__(16) __bf16 Bs[2][128 * LDG];

  const int tid = threadIdx.x;
  const int w = tid >> 5, lane = tid & 31;
  const int grp = lane >> 4, ln = lane & 15;
  const int wm = w >> 1, wn = w & 1;                 // 4x2 wave grid
  const int m0 = blockIdx.y * 128, n0 = blockIdx.x * 128;

  const unsigned int ldsA[2] = {lds_offset(As[0]), lds_offset(As[1])};
  const unsigned int ldsB[2] = {lds_offset(Bs[0]), lds_offset(Bs[1])};
  const __bf16* Ag = A + (size_t)m0 * KTOT;
  const __bf16* Bg = Bt + (size_t)n0 * KTOT;

  v8f acc[2][4];
#pragma unroll
  for (int mi = 0; mi < 2; ++mi)
#pragma unroll
    for (int ni = 0; ni < 4; ++ni) acc[mi][ni] = {};

  constexpr int ITERS = KTOT / 64;
  if (tid < 32) {                                    // preload buffer 0
    tdm_load_2d(ldsA[0], Ag, KTOT, ROWS, KTOT, 64, 128, 4, 3);
    tdm_load_2d(ldsB[0], Bg, KTOT, NTOT, KTOT, 64, 128, 4, 3);
  }

  for (int it = 0; it < ITERS; ++it) {
    if (tid < 32) {
      if (it + 1 < ITERS) {                          // prefetch next buffer
        int k0 = (it + 1) * 64;
        tdm_load_2d(ldsA[(it + 1) & 1], Ag + k0, KTOT, ROWS, KTOT, 64, 128, 4,
                    3);
        tdm_load_2d(ldsB[(it + 1) & 1], Bg + k0, KTOT, NTOT, KTOT, 64, 128, 4,
                    3);
        __builtin_amdgcn_s_wait_tensorcnt(2);        // current buffer done
      } else {
        __builtin_amdgcn_s_wait_tensorcnt(0);
      }
    }
    __syncthreads();

    const __bf16* Ac = As[it & 1];
    const __bf16* Bc = Bs[it & 1];
#pragma unroll
    for (int ks = 0; ks < 2; ++ks) {
      v16bf af[2], bfr[4];
#pragma unroll
      for (int mi = 0; mi < 2; ++mi)
        af[mi] = load_a16x32(Ac, LDG, wm * 32 + mi * 16, ks * 32);
#pragma unroll
      for (int ni = 0; ni < 4; ++ni)
        bfr[ni] = load_bT32x16(Bc, LDG, wn * 64 + ni * 16, ks * 32);
#pragma unroll
      for (int mi = 0; mi < 2; ++mi)
#pragma unroll
        for (int ni = 0; ni < 4; ++ni)
          acc[mi][ni] = wmma_bf16(af[mi], bfr[ni], acc[mi][ni]);
    }
    __syncthreads();
  }

  // Epilogue.  C tile layout: element g -> row = g + 8*grp, col = ln.
#pragma unroll
  for (int mi = 0; mi < 2; ++mi)
#pragma unroll
    for (int ni = 0; ni < 4; ++ni)
#pragma unroll
      for (int g = 0; g < 8; ++g) {
        int row = m0 + wm * 32 + mi * 16 + g + 8 * grp;
        int col = n0 + wn * 64 + ni * 16 + ln;
        float val = acc[mi][ni][g] + bias[col];
        if (MODE == 0) {
          int part = col >> 10;                      // 0:Q 1:K 2:V
          int h = col & 1023, head = h >> 6, d = h & 63;
          int b = row >> 11, s = row & 2047;
          obf[(size_t)part * ((size_t)ROWS * HID) +
              (((size_t)(b * NH + head) * SEQ + s) * HS + d)] = f2bf(val);
        } else {
          of32[(size_t)row * HID + col] = val;
        }
      }
}

// ---------------------------------------------------------------------------
// Kernel 3: flash attention.  Grid (qblock=16, head=16, batch=2), 256 thr.
// Each wave owns 16 query rows; stream 64-key K/V blocks (double-buffered TDM)
// with online softmax.  Row sums come from an extra WMMA against a ones
// matrix (accumulator o[4]) -- no shuffle-tree sum reduction.
// ---------------------------------------------------------------------------
#define LKV 72
#define KB  64

__global__ __launch_bounds__(256) void attn_kernel(
    const __bf16* __restrict__ q, const __bf16* __restrict__ k,
    const __bf16* __restrict__ v, __bf16* __restrict__ av) {
  __shared__ __align__(16) __bf16 Qs[128 * LKV];
  __shared__ __align__(16) __bf16 Ks[2][KB * LKV];
  __shared__ __align__(16) __bf16 Vs[2][KB * LKV];
  __shared__ __align__(16) __bf16 Ps[8 * 16 * KB];

  const int tid = threadIdx.x;
  const int w = tid >> 5, lane = tid & 31;
  const int grp = lane >> 4, ln = lane & 15;
  const int qblk = blockIdx.x, head = blockIdx.y, b = blockIdx.z;

  const size_t hoff = (size_t)(b * NH + head) * SEQ * HS;
  const __bf16* qg = q + hoff + (size_t)qblk * 128 * HS;
  const __bf16* kg = k + hoff;
  const __bf16* vg = v + hoff;

  const unsigned int ldsQ = lds_offset(Qs);
  const unsigned int ldsK[2] = {lds_offset(Ks[0]), lds_offset(Ks[1])};
  const unsigned int ldsV[2] = {lds_offset(Vs[0]), lds_offset(Vs[1])};

  if (tid < 32) {                 // Q tile (128x64) + first K/V block (64x64)
    tdm_load_2d(ldsQ, qg, HS, SEQ, HS, HS, 128, 4, 3);
    tdm_load_2d(ldsK[0], kg, HS, SEQ, HS, HS, KB, 4, 3);
    tdm_load_2d(ldsV[0], vg, HS, SEQ, HS, HS, KB, 4, 3);
    __builtin_amdgcn_s_wait_tensorcnt(2);            // Q complete
  }
  __syncthreads();

  const v16bf aq0 = load_a16x32(Qs, LKV, w * 16, 0);
  const v16bf aq1 = load_a16x32(Qs, LKV, w * 16, 32);

  const __bf16 one = f2bf(1.0f);
  v16bf ones;
#pragma unroll
  for (int j = 0; j < 16; ++j) ones[j] = one;

  float rmax[8];
  v8f o[5];                       // o[0..3] = output, o[4] = row sums
#pragma unroll
  for (int g = 0; g < 8; ++g) rmax[g] = -1e30f;
#pragma unroll
  for (int t = 0; t < 5; ++t) o[t] = {};

  __bf16* pw = Ps + w * (16 * KB);
  constexpr int ITERS = SEQ / KB;

  for (int it = 0; it < ITERS; ++it) {
    if (tid < 32) {
      if (it + 1 < ITERS) {                          // prefetch next K/V
        const size_t off = (size_t)(it + 1) * KB * HS;
        tdm_load_2d(ldsK[(it + 1) & 1], kg + off, HS, SEQ, HS, HS, KB, 4, 3);
        tdm_load_2d(ldsV[(it + 1) & 1], vg + off, HS, SEQ, HS, HS, KB, 4, 3);
        __builtin_amdgcn_s_wait_tensorcnt(2);        // current K/V done
      } else {
        __builtin_amdgcn_s_wait_tensorcnt(0);
      }
    }
    __syncthreads();

    const __bf16* Kc = Ks[it & 1];
    const __bf16* Vc = Vs[it & 1];

    // scores 16x64 : S = Q(16x64) * K^T   (K rows in LDS are exactly B^T)
    v8f s[4];
#pragma unroll
    for (int i = 0; i < 4; ++i) {
      s[i] = {};
      s[i] = wmma_bf16(aq0, load_bT32x16(Kc, LKV, i * 16, 0), s[i]);
      s[i] = wmma_bf16(aq1, load_bT32x16(Kc, LKV, i * 16, 32), s[i]);
    }

    // online softmax per row (row = g + 8*grp within the wave's 16 rows)
#pragma unroll
    for (int g = 0; g < 8; ++g) {
      float x0 = s[0][g] * 0.125f;                   // 1/sqrt(64)
      float x1 = s[1][g] * 0.125f;
      float x2 = s[2][g] * 0.125f;
      float x3 = s[3][g] * 0.125f;
      float m = redmax16(fmaxf(fmaxf(x0, x1), fmaxf(x2, x3)));
      float mn = fmaxf(rmax[g], m);
      float alpha = __expf(rmax[g] - mn);
      rmax[g] = mn;
      o[0][g] *= alpha; o[1][g] *= alpha; o[2][g] *= alpha;
      o[3][g] *= alpha; o[4][g] *= alpha;
      int row = g + 8 * grp;
      pw[row * KB + ln]      = f2bf(__expf(x0 - mn));
      pw[row * KB + 16 + ln] = f2bf(__expf(x1 - mn));
      pw[row * KB + 32 + ln] = f2bf(__expf(x2 - mn));
      pw[row * KB + 48 + ln] = f2bf(__expf(x3 - mn));
    }

    // O += P(16x64) * V(64x64);  rowsum += P * ones
    v16bf ap0 = load_a16x32(pw, KB, 0, 0);
    v16bf ap1 = load_a16x32(pw, KB, 0, 32);
#pragma unroll
    for (int t = 0; t < 4; ++t) {
      o[t] = wmma_bf16(ap0, load_b32x16_rm(Vc, LKV, 0, t * 16), o[t]);
      o[t] = wmma_bf16(ap1, load_b32x16_rm(Vc, LKV, 32, t * 16), o[t]);
    }
    o[4] = wmma_bf16(ap0, ones, o[4]);
    o[4] = wmma_bf16(ap1, ones, o[4]);
    __syncthreads();
  }

  // write av in [B][S][nh][hs] (== [B*S][H]) bf16
#pragma unroll
  for (int g = 0; g < 8; ++g) {
    float inv = 1.0f / o[4][g];
    int row = qblk * 128 + w * 16 + g + 8 * grp;
#pragma unroll
    for (int t = 0; t < 4; ++t) {
      av[((size_t)(b * SEQ + row)) * HID + head * HS + t * 16 + ln] =
          f2bf(o[t][g] * inv);
    }
  }
}

// ---------------------------------------------------------------------------
// launch
// ---------------------------------------------------------------------------
extern "C" void kernel_launch(void* const* d_in, const int* in_sizes, int n_in,
                              void* d_out, int out_size, void* d_ws,
                              size_t ws_size, hipStream_t stream) {
  (void)in_sizes; (void)n_in; (void)out_size; (void)ws_size;
  const float* x     = (const float*)d_in[0];
  const float* W_qkv = (const float*)d_in[1];
  const float* b_qkv = (const float*)d_in[2];
  const float* W_o   = (const float*)d_in[3];
  const float* b_o   = (const float*)d_in[4];
  float* out = (float*)d_out;

  __bf16* ws = (__bf16*)d_ws;
  __bf16* xb    = ws;                                // 4,194,304
  __bf16* wqkvt = xb + (size_t)ROWS * HID;           // 3,145,728
  __bf16* wot   = wqkvt + (size_t)QKVN * HID;        // 1,048,576
  __bf16* qkv   = wot + (size_t)HID * HID;           // 3 * 4,194,304 (Q,K,V)
  __bf16* qb    = qkv;
  __bf16* kb    = qkv + (size_t)ROWS * HID;
  __bf16* vb    = qkv + 2 * (size_t)ROWS * HID;
  __bf16* av    = qkv + 3 * (size_t)ROWS * HID;      // 4,194,304

  convert_kernel<<<(ROWS * HID) / 256, 256, 0, stream>>>(x, W_qkv, W_o, xb,
                                                         wqkvt, wot);

  gemm128_kernel<0, QKVN, HID>
      <<<dim3(QKVN / 128, ROWS / 128), 256, 0, stream>>>(xb, wqkvt, b_qkv, qkv,
                                                         nullptr);

  attn_kernel<<<dim3(SEQ / 128, NH, BATCH), 256, 0, stream>>>(qb, kb, vb, av);

  gemm128_kernel<1, HID, HID>
      <<<dim3(HID / 128, ROWS / 128), 256, 0, stream>>>(av, wot, b_o, nullptr,
                                                        out);
}
